// PCSA_79001628442774
// MI455X (gfx1250) — matrix operand order repaired and checked by reading, not compiled, not run on hardware
//
#include <hip/hip_runtime.h>
#include <hip/hip_bf16.h>
#include <math.h>

// ---------------- problem constants ----------------
#define EPS_F 1e-5f
constexpr int BB  = 32;          // batch
constexpr int CC  = 256;         // channels
constexpr int GRP = 8;           // groups
constexpr int CG  = 32;          // channels per group
constexpr int BG  = 256;         // BB*GRP
constexpr int HH  = 56, WW = 56, HW = HH * WW;

// ---------------- vector types for WMMA ----------------
typedef __attribute__((ext_vector_type(16))) __bf16 v16bf;
typedef __attribute__((ext_vector_type(8)))  __bf16 v8bf;
typedef __attribute__((ext_vector_type(8)))  float  v8f;

__device__ __forceinline__ float sigmoidf_(float v) { return 1.f / (1.f + expf(-v)); }

__device__ __forceinline__ void wait_async0() {
#if __has_builtin(__builtin_amdgcn_s_wait_asynccnt)
  __builtin_amdgcn_s_wait_asynccnt(0);
#else
  asm volatile("s_wait_asynccnt 0x0" ::: "memory");
#endif
}

// async copy of 16 bytes: global (saddr base + voffset) -> LDS[ldsaddr]
__device__ __forceinline__ void async_copy_b128(unsigned lds_addr, unsigned gofs,
                                                const void* sbase) {
  asm volatile("global_load_async_to_lds_b128 %0, %1, %2"
               :: "v"(lds_addr), "v"(gofs), "s"(sbase) : "memory");
}

__device__ __forceinline__ float blockReduceSum256(float v, float* red) {
  int t = threadIdx.x;
  red[t] = v; __syncthreads();
  for (int s = 128; s > 0; s >>= 1) { if (t < s) red[t] += red[t + s]; __syncthreads(); }
  float r = red[0]; __syncthreads(); return r;
}

// ======================================================================
// Weight reorder + bf16 convert:
//   w (OC, IC, 3, 3) f32  ->  wbf [kh][icb][kw][oc(OC)][ic(32)] bf16
// so each conv K-stage (kh,icb) sees 3 contiguous OC*32-element blocks.
// ======================================================================
__global__ __launch_bounds__(256) void k_wreorder(const float* __restrict__ w,
                                                  __bf16* __restrict__ wbf,
                                                  int OC, int IC) {
  size_t total = (size_t)9 * OC * IC;
  size_t i = (size_t)blockIdx.x * 256 + threadIdx.x;
  if (i >= total) return;
  int ic  = (int)(i % 32);
  int oc  = (int)((i / 32) % OC);
  int kw  = (int)((i / (32 * (size_t)OC)) % 3);
  int icb = (int)((i / (96 * (size_t)OC)) % (IC / 32));
  int kh  = (int)( i / (96 * (size_t)OC * (IC / 32)));
  wbf[i] = (__bf16)w[(((size_t)oc * IC + icb * 32 + ic) * 3 + kh) * 3 + kw];
}

// ======================================================================
// Implicit-GEMM 3x3 conv (stride 1, pad 1) via bf16 WMMA, f32 accumulate.
//   in  : (NIMG, IC, 56, 56) f32
//   wbf : reordered bf16 weights (see k_wreorder)
//   out : (NIMG, OC, 56, 56) f32      bias: (OC) f32
// Block: 256 threads = 8 waves; one output row y, 64 px x NTILE ocs.
// Wave w: M-subtile = (w&3)*16 px, N base = (w>>2)*(NTILE/2).
// K loop: stages (kh, ic-chunk-of-32); 3 WMMA K=32 steps (kw) per stage.
// B tiles double-buffered in LDS, filled by global_load_async_to_lds_b128.
// ======================================================================
template <int IC, int OC, int NTILE>
__global__ __launch_bounds__(256) void conv3x3_wmma(
    const float* __restrict__ in, const __bf16* __restrict__ wbf,
    const float* __restrict__ bias, float* __restrict__ out) {
  constexpr int ICC    = 32;
  constexpr int NPW    = NTILE / 32;    // 16x16 N-subtiles per wave (2 or 1)
  constexpr int NICB   = IC / ICC;
  constexpr int NSTAGE = 3 * NICB;      // (kh, icb) stages
  constexpr int CHUNKS = NTILE * 4;     // 16B async chunks per kw block

  const int img = blockIdx.x;
  const int y   = blockIdx.y;
  const int oc0 = blockIdx.z * NTILE;
  const int tid = threadIdx.x, lane = tid & 31, wv = tid >> 5;
  const int msub  = wv & 3;
  const int nbase = (wv >> 2) * (NPW * 16);

  __shared__ __attribute__((aligned(64))) __bf16 sA[66][ICC];           // [x+1][ic]
  __shared__ __attribute__((aligned(64))) __bf16 sB[2][3][NTILE][ICC];  // dbl-buffered

  v8f acc[NPW] = {};
  const float* inImg = in + (size_t)img * IC * HW;

  // Issue async weight-tile copy for stage s into buffer buf.
  auto issueB = [&](int s, int buf) {
    if (tid < CHUNKS) {
      const int kh = s / NICB, icb = s % NICB;
      const unsigned gofs = (unsigned)tid * 16u;
      #pragma unroll
      for (int kw = 0; kw < 3; ++kw) {
        const __bf16* src =
            wbf + (((size_t)(kh * NICB + icb) * 3 + kw) * OC + oc0) * 32;
        unsigned lds = (unsigned)(size_t)(&sB[buf][kw][0][0]) + gofs;
        async_copy_b128(lds, gofs, (const void*)src);
      }
    }
  };

  issueB(0, 0);  // prologue

  for (int s = 0; s < NSTAGE; ++s) {
    const int kh  = s / NICB;
    const int icc = (s % NICB) * ICC;
    const int yi  = y + kh - 1;
    const bool yok = (yi >= 0) && (yi < HH);
    // ---- stage input row (halo, zero-padded) as bf16: sA[xp][ic]
    for (int idx = tid; idx < 66 * ICC; idx += 256) {
      int ic = idx & (ICC - 1);
      int xp = idx >> 5;             // ICC == 32
      int xi = xp - 1;
      float v = 0.f;
      if (yok && xi >= 0 && xi < WW) {
        const float* gp = &inImg[(size_t)(icc + ic) * HW + yi * WW + xi];
        v = *gp;
        if (yi + 1 < HH) __builtin_prefetch(gp + WW, 0, 0);
      }
      sA[xp][ic] = (__bf16)v;
    }
    wait_async0();        // this wave's async B-tile writes complete
    __syncthreads();      // sA + everyone's sB[cur] visible
    if (s + 1 < NSTAGE) issueB(s + 1, (s + 1) & 1);
    // ---- 3 WMMA K-steps (one per kw)
    const int buf  = s & 1;
    const int arow = msub * 16 + (lane & 15);  // pixel x (0..63)
    const int akb  = (lane >> 4) * 16;         // A K-half base
    const int bcol = lane & 15;                // B column within 16
    const int bkb  = (lane >> 4) * 16;         // B K-half base
    #pragma unroll
    for (int kw = 0; kw < 3; ++kw) {
      const __bf16* ap = &sA[arow + kw][akb];
      v8bf alo = *(const v8bf*)ap;             // K = akb..akb+7
      v8bf ahi = *(const v8bf*)(ap + 8);       // K = akb+8..akb+15
      v16bf a;
      #pragma unroll
      for (int i = 0; i < 8; ++i) { a[i] = alo[i]; a[8 + i] = ahi[i]; }
      #pragma unroll
      for (int ns = 0; ns < NPW; ++ns) {
        const __bf16* bp = &sB[buf][kw][nbase + ns * 16 + bcol][bkb];
        v16bf b = *(const v16bf*)bp;           // K = bkb..bkb+15, contiguous
        acc[ns] = __builtin_amdgcn_wmma_f32_16x16x32_bf16(
            false, a, false, b, (short)0, acc[ns], false, false);
      }
    }
    __syncthreads();      // all reads done before next stage overwrites
  }
  // ---- store (C/D layout: vgpr r -> M = r + 8*(lane>>4), N = lane&15)
  const int xb   = msub * 16 + 8 * (lane >> 4);
  const int ccol = lane & 15;
  float* outImg = out + (size_t)img * OC * HW;
  #pragma unroll
  for (int ns = 0; ns < NPW; ++ns) {
    int oc = oc0 + nbase + ns * 16 + ccol;
    float bv = bias[oc];
    #pragma unroll
    for (int r = 0; r < 8; ++r) {
      int x = xb + r;
      if (x < WW) outImg[(size_t)oc * HW + y * WW + x] = acc[ns][r] + bv;
    }
  }
}

// ======================================================================
// Small kernels
// ======================================================================
__global__ void k_zero(float* bsum, float* bsumsq) {
  int t = threadIdx.x;
  if (t < CC) { bsum[t] = 0.f; bsumsq[t] = 0.f; }
}

// Per (bg,cg) channel: row means -> xhw[ch][0..55], col means -> xhw[ch][56..111]
__global__ __launch_bounds__(128) void k_rowcol(const float* __restrict__ x,
                                                float* __restrict__ xhw) {
  int ch = blockIdx.x, t = threadIdx.x;
  const float* src = x + (size_t)ch * HW;
  if (t < 56) {
    float s = 0.f;
    for (int xx = 0; xx < WW; ++xx) s += src[t * WW + xx];
    xhw[(size_t)ch * 112 + t] = s * (1.f / 56.f);
  } else if (t >= 64 && t < 120) {
    int xx = t - 64;
    float s = 0.f;
    for (int yy = 0; yy < HH; ++yy) s += src[yy * WW + xx];
    xhw[(size_t)ch * 112 + 56 + xx] = s * (1.f / 56.f);
  }
}

// 1x1 conv over the 112-length axis + sigmoid: sighw[bg][oc][p]
__global__ __launch_bounds__(128) void k_conv1x1(const float* __restrict__ xhw,
                                                 const float* __restrict__ w1,
                                                 const float* __restrict__ b1,
                                                 float* __restrict__ sighw) {
  __shared__ float sin[CG * 112];
  int bg = blockIdx.x, tid = threadIdx.x;
  for (int i = tid; i < CG * 112; i += 128) sin[i] = xhw[(size_t)bg * CG * 112 + i];
  __syncthreads();
  for (int i = tid; i < CG * 112; i += 128) {
    int oc = i / 112, p = i % 112;
    float s = b1[oc];
    for (int ic = 0; ic < CG; ++ic) s += w1[oc * CG + ic] * sin[ic * 112 + p];
    sighw[(size_t)bg * CG * 112 + i] = sigmoidf_(s);
  }
}

// x0 = x * sig_h[y] * sig_w[x]; per-channel-instance mu/rstd; batch stats atomics
__global__ __launch_bounds__(256) void k_x0_stats(
    const float* __restrict__ x, const float* __restrict__ sighw,
    float* __restrict__ x0, float* __restrict__ mu, float* __restrict__ rstd,
    float* __restrict__ bsum, float* __restrict__ bsumsq) {
  __shared__ float sh[112];
  __shared__ float red[256];
  int ch = blockIdx.x, tid = threadIdx.x;   // ch = b*256 + c
  const float* src = x + (size_t)ch * HW;
  float* dst = x0 + (size_t)ch * HW;
  for (int i = tid; i < 112; i += 256) sh[i] = sighw[(size_t)ch * 112 + i];
  __syncthreads();
  float s = 0.f, sq = 0.f;
  for (int p = tid; p < HW; p += 256) {
    float v = src[p] * sh[p / WW] * sh[56 + p % WW];
    dst[p] = v; s += v; sq += v * v;
  }
  float ts  = blockReduceSum256(s, red);
  float tsq = blockReduceSum256(sq, red);
  if (tid == 0) {
    float m = ts / HW, var = tsq / HW - m * m;
    mu[ch] = m; rstd[ch] = rsqrtf(var + EPS_F);
    atomicAdd(&bsum[ch & (CC - 1)], ts);
    atomicAdd(&bsumsq[ch & (CC - 1)], tsq);
  }
}

__global__ void k_bstats(const float* bsum, const float* bsumsq, float* bm, float* brstd) {
  int c = threadIdx.x;
  if (c < CC) {
    const float invN = 1.f / ((float)BB * HW);
    float m = bsum[c] * invN;
    float var = bsumsq[c] * invN - m * m;
    bm[c] = m; brstd[c] = rsqrtf(var + EPS_F);
  }
}

// per-(bg,cg) sum of conv output x2 (for its softmax-over-channels gate)
__global__ __launch_bounds__(256) void k_chansum(const float* __restrict__ x2,
                                                 float* __restrict__ s2sum) {
  __shared__ float red[256];
  int ch = blockIdx.x;
  const float* src = x2 + (size_t)ch * HW;
  float s = 0.f;
  for (int p = threadIdx.x; p < HW; p += 256) s += src[p];
  float t = blockReduceSum256(s, red);
  if (threadIdx.x == 0) s2sum[ch] = t;
}

// weights1[bg,p] = sum_c x11[c]*x2[c,p] + x21[c]*x1[c,p]; x1 recomputed from x0.
// x1.mean over (h,w) == gn_b exactly -> x11 = softmax(gn_b).
__global__ __launch_bounds__(256) void k_weights1(
    const float* __restrict__ x0, const float* __restrict__ x2,
    const float* __restrict__ mu, const float* __restrict__ rstd,
    const float* __restrict__ gnw, const float* __restrict__ gnb,
    const float* __restrict__ s2sum, float* __restrict__ sigw1) {
  __shared__ float alpha[CG], x11s[CG], k0s;
  int bg = blockIdx.x, tid = threadIdx.x;
  if (tid == 0) {
    float m1 = -1e30f, m2 = -1e30f;
    for (int c = 0; c < CG; ++c) {
      m1 = fmaxf(m1, gnb[c]);
      m2 = fmaxf(m2, s2sum[bg * CG + c] * (1.f / HW));
    }
    float z1 = 0.f, z2 = 0.f;
    for (int c = 0; c < CG; ++c) {
      z1 += expf(gnb[c] - m1);
      z2 += expf(s2sum[bg * CG + c] * (1.f / HW) - m2);
    }
    float k0 = 0.f;
    for (int c = 0; c < CG; ++c) {
      float x11 = expf(gnb[c] - m1) / z1;
      float x21 = expf(s2sum[bg * CG + c] * (1.f / HW) - m2) / z2;
      float r = rstd[bg * CG + c], m = mu[bg * CG + c];
      x11s[c]  = x11;
      alpha[c] = x21 * gnw[c] * r;
      k0 += x21 * (gnb[c] - m * r * gnw[c]);
    }
    k0s = k0;
  }
  __syncthreads();
  const float* p0 = x0 + (size_t)bg * CG * HW;
  const float* p2 = x2 + (size_t)bg * CG * HW;
  for (int p = tid; p < HW; p += 256) {
    float acc = k0s;
    #pragma unroll 4
    for (int c = 0; c < CG; ++c)
      acc += alpha[c] * p0[(size_t)c * HW + p] + x11s[c] * p2[(size_t)c * HW + p];
    sigw1[(size_t)bg * HW + p] = sigmoidf_(acc);
  }
}

// m3[b,p] = mean_c (x0-bm)*brstd ; m4[b,p] = mean_c x4
__global__ __launch_bounds__(256) void k_colmeans(
    const float* __restrict__ x0, const float* __restrict__ x4,
    const float* __restrict__ bm, const float* __restrict__ brstd,
    float* __restrict__ m3, float* __restrict__ m4) {
  int b = blockIdx.y;
  int p = blockIdx.x * 256 + threadIdx.x;
  if (p >= HW) return;
  const float* p0 = x0 + (size_t)b * CC * HW + p;
  const float* p4 = x4 + (size_t)b * CC * HW + p;
  float s3 = 0.f, s4 = 0.f;
  for (int c = 0; c < CC; ++c) {
    s3 += (p0[(size_t)c * HW] - bm[c]) * brstd[c];
    s4 += p4[(size_t)c * HW];
  }
  m3[(size_t)b * HW + p] = s3 * (1.f / CC);
  m4[(size_t)b * HW + p] = s4 * (1.f / CC);
}

// softmax over 3136 pixels (LDS-resident), blockIdx.y selects m3->a3 / m4->a4
__global__ __launch_bounds__(256) void k_softmax_hw(
    const float* __restrict__ m3, const float* __restrict__ m4,
    float* __restrict__ a3, float* __restrict__ a4) {
  __shared__ float buf[HW];
  __shared__ float red[256];
  int b = blockIdx.x, tid = threadIdx.x;
  const float* src = (blockIdx.y == 0 ? m3 : m4) + (size_t)b * HW;
  float* dst = (blockIdx.y == 0 ? a3 : a4) + (size_t)b * HW;
  float mx = -1e30f;
  for (int p = tid; p < HW; p += 256) { float v = src[p]; buf[p] = v; mx = fmaxf(mx, v); }
  red[tid] = mx; __syncthreads();
  for (int s = 128; s > 0; s >>= 1) { if (tid < s) red[tid] = fmaxf(red[tid], red[tid + s]); __syncthreads(); }
  mx = red[0]; __syncthreads();
  float z = 0.f;
  for (int p = tid; p < HW; p += 256) { float e = expf(buf[p] - mx); buf[p] = e; z += e; }
  z = blockReduceSum256(z, red);
  float inv = 1.f / z;
  for (int p = tid; p < HW; p += 256) dst[p] = buf[p] * inv;
}

// weights2[b,c] = sum_p a3*x4 + a4*x3 ; x3 folded (sum_p a4 == 1)
__global__ __launch_bounds__(256) void k_weights2(
    const float* __restrict__ x0, const float* __restrict__ x4,
    const float* __restrict__ a3, const float* __restrict__ a4,
    const float* __restrict__ bm, const float* __restrict__ brstd,
    float* __restrict__ sw2) {
  __shared__ float red[256];
  int c = blockIdx.x, b = blockIdx.y;
  const float* p0  = x0 + ((size_t)b * CC + c) * HW;
  const float* p4  = x4 + ((size_t)b * CC + c) * HW;
  const float* pa3 = a3 + (size_t)b * HW;
  const float* pa4 = a4 + (size_t)b * HW;
  float r = brstd[c], s = 0.f;
  for (int p = threadIdx.x; p < HW; p += 256)
    s += pa3[p] * p4[p] + pa4[p] * p0[p] * r;
  float t = blockReduceSum256(s, red);
  if (threadIdx.x == 0) sw2[b * CC + c] = sigmoidf_(t - bm[c] * r);
}

// out = x * (sigmoid(weights1) + sigmoid(weights2))   (gx reshaped == x)
__global__ __launch_bounds__(256) void k_final(
    const float* __restrict__ x, const float* __restrict__ sigw1,
    const float* __restrict__ sw2, float* __restrict__ out) {
  size_t idx = (size_t)blockIdx.x * 256 + threadIdx.x;
  if (idx >= (size_t)BB * CC * HW) return;
  int p = (int)(idx % HW);
  size_t bc = idx / HW;
  int c = (int)(bc % CC), b = (int)(bc / CC);
  int bg = b * GRP + c / CG;
  out[idx] = x[idx] * (sigw1[(size_t)bg * HW + p] + sw2[bc]);
}

// ======================================================================
extern "C" void kernel_launch(void* const* d_in, const int* in_sizes, int n_in,
                              void* d_out, int out_size, void* d_ws, size_t ws_size,
                              hipStream_t stream) {
  (void)in_sizes; (void)n_in; (void)out_size; (void)ws_size;
  const float* x   = (const float*)d_in[0];
  const float* w1  = (const float*)d_in[1];
  const float* b1  = (const float*)d_in[2];
  const float* w3a = (const float*)d_in[3];
  const float* b3a = (const float*)d_in[4];
  const float* w3b = (const float*)d_in[5];
  const float* b3b = (const float*)d_in[6];
  const float* gnw = (const float*)d_in[7];
  const float* gnb = (const float*)d_in[8];
  float* out = (float*)d_out;

  float* w = (float*)d_ws;
  size_t off = 0;
  auto alloc = [&](size_t n) { n = (n + 3) & ~(size_t)3; float* p = w + off; off += n; return p; };
  float* x0     = alloc((size_t)BG * CG * HW);   // (b,c,p) flat
  float* xc     = alloc((size_t)BG * CG * HW);   // x2 then x4 (disjoint lifetimes)
  float* xhw    = alloc((size_t)BG * CG * 112);
  float* sighw  = alloc((size_t)BG * CG * 112);
  float* mu     = alloc(BG * CG);
  float* rstd   = alloc(BG * CG);
  float* bsum   = alloc(CC);
  float* bsumsq = alloc(CC);
  float* bm     = alloc(CC);
  float* brstd  = alloc(CC);
  float* s2sum  = alloc(BG * CG);
  float* sigw1  = alloc((size_t)BG * HW);
  float* m3     = alloc((size_t)BB * HW);
  float* m4     = alloc((size_t)BB * HW);
  float* a3     = alloc((size_t)BB * HW);
  float* a4     = alloc((size_t)BB * HW);
  float* sw2    = alloc(BB * CC);
  __bf16* wbfA  = (__bf16*)alloc((size_t)9 * CG * CG / 2);   // 9216 halves
  __bf16* wbfB  = (__bf16*)alloc((size_t)9 * CC * CC / 2);   // 589824 halves

  k_zero<<<1, 256, 0, stream>>>(bsum, bsumsq);
  k_wreorder<<<(9 * CG * CG + 255) / 256, 256, 0, stream>>>(w3a, wbfA, CG, CG);
  k_wreorder<<<(9 * CC * CC + 255) / 256, 256, 0, stream>>>(w3b, wbfB, CC, CC);
  k_rowcol<<<BG * CG, 128, 0, stream>>>(x, xhw);
  k_conv1x1<<<BG, 128, 0, stream>>>(xhw, w1, b1, sighw);
  k_x0_stats<<<BG * CG, 256, 0, stream>>>(x, sighw, x0, mu, rstd, bsum, bsumsq);
  k_bstats<<<1, 256, 0, stream>>>(bsum, bsumsq, bm, brstd);
  // grouped 3x3 conv: 256 images of 32ch  -> x2 (in xc)
  conv3x3_wmma<32, 32, 32><<<dim3(BG, HH, 1), 256, 0, stream>>>(x, wbfA, b3a, xc);
  k_chansum<<<BG * CG, 256, 0, stream>>>(xc, s2sum);
  k_weights1<<<BG, 256, 0, stream>>>(x0, xc, mu, rstd, gnw, gnb, s2sum, sigw1);
  // full 3x3 conv: 32 images of 256ch -> x4 (reuses xc)
  conv3x3_wmma<256, 256, 64><<<dim3(BB, HH, 4), 256, 0, stream>>>(x, wbfB, b3b, xc);
  k_colmeans<<<dim3((HW + 255) / 256, BB), 256, 0, stream>>>(x0, xc, bm, brstd, m3, m4);
  k_softmax_hw<<<dim3(BB, 2), 256, 0, stream>>>(m3, m4, a3, a4);
  k_weights2<<<dim3(CC, BB), 256, 0, stream>>>(x0, xc, a3, a4, bm, brstd, sw2);
  k_final<<<(int)(((size_t)BB * CC * HW + 255) / 256), 256, 0, stream>>>(x, sigw1, sw2, out);
}
